// SingleHeadAttention_41248865911168
// MI455X (gfx1250) — matrix operand order repaired and checked by reading, not compiled, not run on hardware
//
#include <hip/hip_runtime.h>

// ---------------------------------------------------------------------------
// Single-head causal attention for MI455X (gfx1250, wave32, WMMA).
//   x: [8, 2048, 1024] f32, Wq/Wk/Wv: [1024, 64] f32 -> out [8, 2048, 64] f32
//
// Kernel 0: weight prep    -- W[e][h] f32 -> Wt[h][e] f16 (contiguous B-frags)
// Kernel 1: QKV projection -- v_wmma_f32_16x16x32_f16; Q,K stored [b][t][h] f16,
//                             V stored transposed [b][h][t] f16.
// Kernel 2: flash attention-- online softmax, WMMA Q.K^T and P.V, LDS bounce
//                             to re-stripe P from C-layout to A-layout.
// All WMMA B operands are contiguous 32B-per-lane loads (global_load_b128 pairs).
// ---------------------------------------------------------------------------

typedef __attribute__((ext_vector_type(16))) _Float16 v16h;
typedef __attribute__((ext_vector_type(8)))  _Float16 v8h;
typedef __attribute__((ext_vector_type(8)))  float    v8f;

constexpr int kB    = 8;
constexpr int kT    = 2048;
constexpr int kEmb  = 1024;
constexpr int kHead = 64;

union AFrag { v16h v; _Float16 h[16]; };
union H8    { v8h  v; _Float16 h[8];  };

#define WMMA_F16(a, b, c) \
    __builtin_amdgcn_wmma_f32_16x16x32_f16(false, (a), false, (b), (short)0, (c), false, false)

// Reductions across a 16-lane half of the wave (xor masks 1..8 stay in-half).
__device__ __forceinline__ float half_max(float v) {
#pragma unroll
    for (int mm = 8; mm >= 1; mm >>= 1) v = fmaxf(v, __shfl_xor(v, mm, 32));
    return v;
}
__device__ __forceinline__ float half_sum(float v) {
#pragma unroll
    for (int mm = 8; mm >= 1; mm >>= 1) v += __shfl_xor(v, mm, 32);
    return v;
}

// ---------------------------------------------------------------------------
// Kernel 0: Wt[h][e] = (f16) W[e][h]  for q,k,v.  64*1024 elems per matrix.
// ---------------------------------------------------------------------------
__global__ __launch_bounds__(256)
void weight_prep_kernel(const float* __restrict__ Wq,
                        const float* __restrict__ Wk,
                        const float* __restrict__ Wv,
                        _Float16* __restrict__ wtq,
                        _Float16* __restrict__ wtk,
                        _Float16* __restrict__ wtv)
{
    const int idx = blockIdx.x * 256 + threadIdx.x;   // [0, 64*1024)
    const int h = idx >> 10;                          // head column
    const int e = idx & (kEmb - 1);                   // emb row
    const size_t src = (size_t)e * kHead + h;
    wtq[idx] = (_Float16)Wq[src];
    wtk[idx] = (_Float16)Wk[src];
    wtv[idx] = (_Float16)Wv[src];
}

// ---------------------------------------------------------------------------
// Kernel 1: q/k/v = x @ W.  Block = 128 threads (4 waves), 16 tokens/block.
// Wave w owns head columns [16w, 16w+16).
// ---------------------------------------------------------------------------
__global__ __launch_bounds__(128)
void qkv_proj_kernel(const float* __restrict__ x,
                     const _Float16* __restrict__ wtq,
                     const _Float16* __restrict__ wtk,
                     const _Float16* __restrict__ wtv,
                     _Float16* __restrict__ q,      // [b][t][h]
                     _Float16* __restrict__ k,      // [b][t][h]
                     _Float16* __restrict__ vt)     // [b][h][t]  (transposed!)
{
    const int lane = threadIdx.x & 31;
    const int wave = threadIdx.x >> 5;
    const int b    = blockIdx.x / (kT / 16);
    const int t0   = (blockIdx.x % (kT / 16)) * 16;

    const int m  = lane & 15;   // A-fragment row (token)
    const int kh = lane >> 4;   // lane half -> K sub-range
    const int n  = lane & 15;   // B/C/D column within tile
    const int nc = wave * 16 + n;

    const size_t xrow = ((size_t)b * kT + t0 + m) * kEmb;
    const size_t wrow = (size_t)nc * kEmb + kh * 16;   // Wt row for this lane

    v8f cq = {}, ck = {}, cv = {};

    for (int k0 = 0; k0 < kEmb; k0 += 32) {
        // A: x[t0+m, k0 .. k0+31]  (f32 -> f16, coalesced b128 loads)
        AFrag a;
#pragma unroll
        for (int j = 0; j < 8; ++j) {
            a.h[j]     = (_Float16)x[xrow + k0 + kh * 16 + j];
            a.h[8 + j] = (_Float16)x[xrow + k0 + kh * 16 + 8 + j];
        }
        // B: Wt[nc, k0+kh*16 .. +15] -- 16 contiguous halves per lane (32B)
        const v16h bq = *(const v16h*)(wtq + wrow + k0);
        const v16h bk = *(const v16h*)(wtk + wrow + k0);
        const v16h bv = *(const v16h*)(wtv + wrow + k0);

        if (k0 + 32 < kEmb)
            __builtin_prefetch(&x[xrow + k0 + 32], 0, 0);   // global_prefetch_b8

        cq = WMMA_F16(a.v, bq, cq);
        ck = WMMA_F16(a.v, bk, ck);
        cv = WMMA_F16(a.v, bv, cv);
    }

    // Q,K: D layout (r,lane) -> row t0+r+8*kh, col nc  (token-major)
#pragma unroll
    for (int r = 0; r < 8; ++r) {
        const size_t row = ((size_t)b * kT + t0 + r + 8 * kh) * kHead;
        q[row + nc] = (_Float16)cq[r];
        k[row + nc] = (_Float16)ck[r];
    }
    // V: transposed store -- lane holds 8 consecutive tokens for head nc.
    H8 vo;
#pragma unroll
    for (int r = 0; r < 8; ++r) vo.h[r] = (_Float16)cv[r];
    *(v8h*)(vt + ((size_t)b * kHead + nc) * kT + t0 + 8 * kh) = vo.v;
}

// ---------------------------------------------------------------------------
// Kernel 2: flash attention, one wave per 16-query tile, 32 s-cols per step.
// ---------------------------------------------------------------------------
__global__ __launch_bounds__(32)
void flash_attn_kernel(const _Float16* __restrict__ q,     // [b][t][h]
                       const _Float16* __restrict__ kmat,  // [b][t][h]
                       const _Float16* __restrict__ vt,    // [b][h][t]
                       float* __restrict__ out)
{
    __shared__ _Float16 pbuf[16 * 32];   // P tile: C-layout -> A-layout bounce

    const int lane = threadIdx.x & 31;
    const int b    = blockIdx.x / (kT / 16);
    const int t0   = (blockIdx.x % (kT / 16)) * 16;

    const int m  = lane & 15;
    const int kh = lane >> 4;
    const int n  = lane & 15;

    // Q fragments: head dims [0,32) and [32,64)
    const size_t qrow = ((size_t)b * kT + t0 + m) * kHead;
    AFrag qa0, qa1;
#pragma unroll
    for (int j = 0; j < 8; ++j) {
        qa0.h[j]     = q[qrow + kh * 16 + j];
        qa0.h[8 + j] = q[qrow + kh * 16 + 8 + j];
        qa1.h[j]     = q[qrow + 32 + kh * 16 + j];
        qa1.h[8 + j] = q[qrow + 32 + kh * 16 + 8 + j];
    }

    float rowmax[8], rowsum[8];
#pragma unroll
    for (int r = 0; r < 8; ++r) { rowmax[r] = -1e30f; rowsum[r] = 0.0f; }
    v8f o0 = {}, o1 = {}, o2 = {}, o3 = {};

    const float scale = 0.03125f;   // 1/sqrt(1024) -- reference scales by emb
    const size_t vbb = (size_t)b * kHead * kT;

    for (int s0 = 0; s0 < t0 + 16; s0 += 32) {
        // --- scores: K^T B-fragments; column n of B == row (s0+n) of K ---
        const int s1i = min(s0 + n, kT - 1);        // clamp OOB (masked anyway)
        const int s2i = min(s0 + 16 + n, kT - 1);
        const _Float16* kp1 = kmat + ((size_t)b * kT + s1i) * kHead + kh * 16;
        const _Float16* kp2 = kmat + ((size_t)b * kT + s2i) * kHead + kh * 16;
        const v16h kb10 = *(const v16h*)(kp1);        // h chunk [0,32)
        const v16h kb11 = *(const v16h*)(kp1 + 32);   // h chunk [32,64)
        const v16h kb20 = *(const v16h*)(kp2);
        const v16h kb21 = *(const v16h*)(kp2 + 32);

        v8f s1 = {}, s2 = {};
        s1 = WMMA_F16(qa0.v, kb10, s1);
        s1 = WMMA_F16(qa1.v, kb11, s1);
        s2 = WMMA_F16(qa0.v, kb20, s2);
        s2 = WMMA_F16(qa1.v, kb21, s2);

        // --- online softmax (row stats live per lane-half, replicated) ---
        const bool needMask = (s0 + 31 > t0);   // uniform across the wave
        float p1[8], p2[8];
#pragma unroll
        for (int r = 0; r < 8; ++r) {
            const int trow = t0 + r + 8 * kh;
            float v1 = s1[r] * scale;
            float v2 = s2[r] * scale;
            if (needMask) {
                if (s0 + n      > trow) v1 = -1e30f;
                if (s0 + 16 + n > trow) v2 = -1e30f;
            }
            const float tmax = half_max(fmaxf(v1, v2));
            const float nm   = fmaxf(rowmax[r], tmax);
            const float corr = __expf(rowmax[r] - nm);
            rowmax[r] = nm;
            const float e1 = __expf(v1 - nm);
            const float e2 = __expf(v2 - nm);
            rowsum[r] = rowsum[r] * corr + half_sum(e1 + e2);
            o0[r] *= corr; o1[r] *= corr; o2[r] *= corr; o3[r] *= corr;
            p1[r] = e1; p2[r] = e2;
        }

        // --- re-stripe P (C-layout -> A-layout) through LDS ---
#pragma unroll
        for (int r = 0; r < 8; ++r) {
            const int row = r + 8 * kh;
            pbuf[row * 32 + n]      = (_Float16)p1[r];
            pbuf[row * 32 + 16 + n] = (_Float16)p2[r];
        }
        __syncthreads();
        AFrag pa;
#pragma unroll
        for (int j = 0; j < 8; ++j) {
            pa.h[j]     = pbuf[m * 32 + kh * 16 + j];
            pa.h[8 + j] = pbuf[m * 32 + kh * 16 + 8 + j];
        }
        __syncthreads();

        // --- V B-fragments from transposed V: 16 contiguous halves/lane ---
        const int sv = min(s0 + kh * 16, kT - 16);   // clamp tail chunk (P==0)
        const _Float16* vp = vt + vbb + sv;
        const v16h vb0 = *(const v16h*)(vp + (size_t)(n)      * kT);
        const v16h vb1 = *(const v16h*)(vp + (size_t)(16 + n) * kT);
        const v16h vb2 = *(const v16h*)(vp + (size_t)(32 + n) * kT);
        const v16h vb3 = *(const v16h*)(vp + (size_t)(48 + n) * kT);

        o0 = WMMA_F16(pa.v, vb0, o0);
        o1 = WMMA_F16(pa.v, vb1, o1);
        o2 = WMMA_F16(pa.v, vb2, o2);
        o3 = WMMA_F16(pa.v, vb3, o3);
    }

    // Normalize and store f32 output.
#pragma unroll
    for (int r = 0; r < 8; ++r) {
        const float inv = 1.0f / rowsum[r];
        const size_t orow = ((size_t)b * kT + t0 + r + 8 * kh) * kHead;
        out[orow + n]      = o0[r] * inv;
        out[orow + 16 + n] = o1[r] * inv;
        out[orow + 32 + n] = o2[r] * inv;
        out[orow + 48 + n] = o3[r] * inv;
    }
}

// ---------------------------------------------------------------------------
extern "C" void kernel_launch(void* const* d_in, const int* in_sizes, int n_in,
                              void* d_out, int out_size, void* d_ws, size_t ws_size,
                              hipStream_t stream)
{
    const float* x  = (const float*)d_in[0];
    const float* Wq = (const float*)d_in[1];
    const float* Wk = (const float*)d_in[2];
    const float* Wv = (const float*)d_in[3];
    float* out = (float*)d_out;

    const size_t nqkv = (size_t)kB * kT * kHead;    // 1,048,576 halves each
    const size_t nwt  = (size_t)kHead * kEmb;       //    65,536 halves each
    _Float16* qws = (_Float16*)d_ws;
    _Float16* kws = qws + nqkv;
    _Float16* vws = kws + nqkv;                     // transposed V
    _Float16* wtq = vws + nqkv;
    _Float16* wtk = wtq + nwt;
    _Float16* wtv = wtk + nwt;                      // total ~6.7 MB of ws

    weight_prep_kernel<<<dim3(kHead * kEmb / 256), 256, 0, stream>>>(
        Wq, Wk, Wv, wtq, wtk, wtv);

    dim3 grid(kB * kT / 16);   // 1024 tiles of 16 tokens
    qkv_proj_kernel<<<grid, 128, 0, stream>>>(x, wtq, wtk, wtv, qws, kws, vws);
    flash_attn_kernel<<<grid, 32, 0, stream>>>(qws, kws, vws, out);
}
